// AttentionBlock_16398185136213
// MI455X (gfx1250) — matrix operand order repaired and checked by reading, not compiled, not run on hardware
//
#include <hip/hip_runtime.h>

// ---------------------------------------------------------------------------
// Sizes (fixed by the reference)
// ---------------------------------------------------------------------------
#define BB 8
#define NN 1024
#define DIN 512
#define HH 8
#define DHEAD 64
#define DEXP 2048
#define DQ 512            // H*DHEAD
#define DV 4096           // H*DIN
#define DQKV 5120         // 2*DQ + DV
#define MROWS (BB * NN)   // 8192

typedef __attribute__((ext_vector_type(16))) __bf16 v16bf;
typedef __attribute__((ext_vector_type(8)))  float  v8f;
typedef __attribute__((ext_vector_type(4)))  unsigned u32x4;
typedef __attribute__((ext_vector_type(4)))  int      i32x4;

union Frag {
  v16bf v;
  unsigned u[8];
};

static __device__ __forceinline__ unsigned short f2bf(float f) {
  union { float f; unsigned u; } c; c.f = f;
  unsigned r = c.u + 0x7FFFu + ((c.u >> 16) & 1u);   // round-to-nearest-even
  return (unsigned short)(r >> 16);
}
static __device__ __forceinline__ float bf2f(unsigned short h) {
  union { unsigned u; float f; } c; c.u = ((unsigned)h) << 16;
  return c.f;
}

// A-fragment K offset (16-bit A 16x32, ISA 7.12.2): VGPR v holds K pair
static __device__ __forceinline__ int koA(int v, int hl) {
  return ((v < 4) ? (2 * v) : (2 * v + 8)) + 8 * hl;
}
// Contiguous K offset used for both A and B in attn_kernel: sequential K, half selects +16
static __device__ __forceinline__ int koB(int v, int hl) {
  return 2 * v + 16 * hl;
}

// ---------------------------------------------------------------------------
// Async global->LDS copy (CDNA5): 16 bytes per lane, ASYNCcnt-tracked.
// Builtin signature (from hipcc diagnostic): arg1 = int4 in addrspace(1).
// Falls back to load + ds_store_b128 if the builtin is unavailable.
// ---------------------------------------------------------------------------
#if defined(__has_builtin)
#if __has_builtin(__builtin_amdgcn_global_load_async_to_lds_b128)
#define HAS_ASYNC_LDS 1
#endif
#endif
#ifndef HAS_ASYNC_LDS
#define HAS_ASYNC_LDS 0
#endif

static __device__ __forceinline__ void cp16_to_lds(const unsigned short* g,
                                                   unsigned short* l) {
#if HAS_ASYNC_LDS
  __builtin_amdgcn_global_load_async_to_lds_b128(
      (__attribute__((address_space(1))) i32x4*)g,
      (__attribute__((address_space(3))) i32x4*)l, 0, 0);
#else
  *(u32x4*)l = *(const u32x4*)g;
#endif
}

static __device__ __forceinline__ void async_wait0() {
#if HAS_ASYNC_LDS
#if __has_builtin(__builtin_amdgcn_s_wait_asynccnt)
  __builtin_amdgcn_s_wait_asynccnt(0);
#else
  asm volatile("s_wait_asynccnt 0x0" ::: "memory");
#endif
#endif
}

// ---------------------------------------------------------------------------
// FeatureNorm over sequence dim (axis=1) + optional pos_enc + optional swish,
// fp32 in -> bf16 out.  One block per (batch, feature) column of 1024 elems.
// ---------------------------------------------------------------------------
__global__ __launch_bounds__(256)
void featnorm_kernel(const float* __restrict__ in, const float* __restrict__ g,
                     const float* __restrict__ be, const float* __restrict__ pos,
                     unsigned short* __restrict__ out, int D, int doSwish) {
  __shared__ float r1[256];
  __shared__ float r2[256];
  const int col = blockIdx.x;
  const int b = col / D;
  const int d = col - b * D;
  const float* base = in + (size_t)b * NN * D + d;
  float s = 0.f, s2 = 0.f;
  for (int n = threadIdx.x; n < NN; n += 256) {
    float v = base[(size_t)n * D];
    s += v; s2 += v * v;
  }
  r1[threadIdx.x] = s; r2[threadIdx.x] = s2;
  __syncthreads();
  for (int st = 128; st > 0; st >>= 1) {
    if (threadIdx.x < st) {
      r1[threadIdx.x] += r1[threadIdx.x + st];
      r2[threadIdx.x] += r2[threadIdx.x + st];
    }
    __syncthreads();
  }
  const float mean = r1[0] * (1.0f / NN);
  const float var  = r2[0] * (1.0f / NN) - mean * mean;
  const float rstd = rsqrtf(var + 1e-5f);
  const float gg = g[d], bb = be[d];
  for (int n = threadIdx.x; n < NN; n += 256) {
    float v = gg * (base[(size_t)n * D] - mean) * rstd + bb;
    if (pos) v += pos[(size_t)n * D + d];
    if (doSwish) v = v / (1.0f + __expf(-v));
    out[(size_t)(b * NN + n) * D + d] = f2bf(v);
  }
}

// ---------------------------------------------------------------------------
// Weight convert + transpose: fp32 KxN row-major -> bf16 NxK row-major
// ---------------------------------------------------------------------------
__global__ __launch_bounds__(256)
void wtrans_kernel(const float* __restrict__ in, unsigned short* __restrict__ out,
                   int K, int N) {
  int tid = blockIdx.x * 256 + threadIdx.x;
  if (tid >= K * N) return;
  int k = tid / N;
  int n = tid - k * N;
  out[(size_t)n * K + k] = f2bf(in[tid]);
}

// ---------------------------------------------------------------------------
// bf16 WMMA GEMM: C(MxN) = A(MxK, bf16 rowmajor) * Bt(NxK, bf16 rowmajor)^T
//                 + bias + residual.  Output fp32 (Cf) or bf16 (Cb).
// Block: 256 threads = 8 waves.  Block tile 128x128.  Each wave: 16x128 strip.
// B tile (128 x 32 K) staged in LDS, double-buffered, async-copied; shared by
// all 8 waves (8x global-traffic reduction on B).  A loaded direct (no reuse
// possible across waves).
// ---------------------------------------------------------------------------
#define BPITCH 40   // 32 K + 8 pad (u16): bank stride 20 dwords -> conflict-free
__global__ __launch_bounds__(256)
void gemm_bf16_kernel(const unsigned short* __restrict__ A,
                      const unsigned short* __restrict__ Bt,
                      const float* __restrict__ bias,
                      const float* __restrict__ resid,
                      float* __restrict__ Cf,
                      unsigned short* __restrict__ Cb,
                      int M, int N, int K) {
  __shared__ unsigned short Bbuf[2][128 * BPITCH];
  const int tid = threadIdx.x;
  const int wave = tid >> 5;
  const int lane = tid & 31;
  const int lr = lane & 15;
  const int hl = lane >> 4;
  const int m_blk = blockIdx.y * 128;
  const int n_blk = blockIdx.x * 128;
  const int m0 = m_blk + wave * 16;

  v8f z = {0.f, 0.f, 0.f, 0.f, 0.f, 0.f, 0.f, 0.f};
  v8f acc[8] = {z, z, z, z, z, z, z, z};

  const unsigned short* arow = A + (size_t)(m0 + lr) * K + 8 * hl;

  // stage one 128x32 B chunk (k0..k0+31) into Bbuf[s]; 512 x 16B transfers
  auto stage = [&](int s, int k0) {
#pragma unroll
    for (int i = 0; i < 2; ++i) {
      const int idx = i * 256 + tid;          // 0..511
      const int row = idx >> 2;               // 0..127
      const int seg = (idx & 3) * 8;          // 0,8,16,24 (u16 elements)
      cp16_to_lds(Bt + (size_t)(n_blk + row) * K + k0 + seg,
                  &Bbuf[s][row * BPITCH + seg]);
    }
  };

  stage(0, 0);
  const int nch = K >> 5;
  for (int c = 0; c < nch; ++c) {
    async_wait0();        // own async transfers for chunk c complete
    __syncthreads();      // whole tile visible; previous buffer free
    if (c + 1 < nch) stage((c + 1) & 1, (c + 1) << 5);
    const int s = c & 1;

    Frag a;
#pragma unroll
    for (int v = 0; v < 8; ++v)
      a.u[v] = *(const unsigned*)(arow + (c << 5) + ((v < 4) ? 2 * v : 2 * v + 8));

#pragma unroll
    for (int t = 0; t < 8; ++t) {
      Frag bfr;
      const unsigned short* bp = &Bbuf[s][(t * 16 + lr) * BPITCH + 8 * hl];
#pragma unroll
      for (int v = 0; v < 8; ++v)
        bfr.u[v] = *(const unsigned*)(bp + ((v < 4) ? 2 * v : 2 * v + 8));
      acc[t] = __builtin_amdgcn_wmma_f32_16x16x32_bf16(
          false, a.v, false, bfr.v, (short)0, acc[t], false, false);
    }
  }

#pragma unroll
  for (int t = 0; t < 8; ++t) {
    const int col = n_blk + t * 16 + lr;
    const float bv = bias ? bias[col] : 0.f;
#pragma unroll
    for (int r = 0; r < 8; ++r) {
      const int row = m0 + r + 8 * hl;
      float val = acc[t][r] + bv;
      if (resid) val += resid[(size_t)row * N + col];
      if (Cf) Cf[(size_t)row * N + col] = val;
      else    Cb[(size_t)row * N + col] = f2bf(val);
    }
  }
}

// ---------------------------------------------------------------------------
// |q|^2 and |k|^2 per (b,h,n) from the bf16 qkv slab
// ---------------------------------------------------------------------------
__global__ __launch_bounds__(256)
void qk2_kernel(const unsigned short* __restrict__ qkv,
                float* __restrict__ q2, float* __restrict__ k2) {
  const int tid = blockIdx.x * 256 + threadIdx.x;   // B*H*N = 65536
  const int bh = tid >> 10;
  const int n = tid & (NN - 1);
  const int b = bh >> 3;
  const int h = bh & 7;
  const unsigned short* p = qkv + (size_t)(b * NN + n) * DQKV + h * DHEAD;
  float sq = 0.f, sk = 0.f;
  for (int c = 0; c < DHEAD; ++c) {
    float qv = bf2f(p[c]);        sq += qv * qv;
    float kv = bf2f(p[DQ + c]);   sk += kv * kv;
  }
  q2[tid] = sq;
  k2[tid] = sk;
}

// ---------------------------------------------------------------------------
// RBF attention: S = exp(-(|q|^2+|k|^2-2qk)/scale^2);  out = S @ V
// One block (4 waves) per (b, h, 16-row i-tile).  S kept in LDS as bf16.
// Output written head-major bf16: out[(b*N+n)*4096 + h*512 + c]
// ---------------------------------------------------------------------------
#define S_PITCH 1032
__global__ __launch_bounds__(128)
void attn_kernel(const unsigned short* __restrict__ qkv,
                 const float* __restrict__ q2,
                 const float* __restrict__ k2,
                 unsigned short* __restrict__ outm,
                 const float* __restrict__ scale_p) {
  __shared__ unsigned short S[16][S_PITCH];
  const int tid = threadIdx.x;
  const int wave = tid >> 5;
  const int lane = tid & 31;
  const int lr = lane & 15;
  const int hl = lane >> 4;
  const int blk = blockIdx.x;          // B*H*(N/16) = 4096
  const int it = blk & 63;
  const int bh = blk >> 6;
  const int b = bh >> 3;
  const int h = bh & 7;
  const int i0 = it * 16;
  const float sc = scale_p[0];
  const float inv_s2 = 1.0f / (sc * sc);

  v8f z = {0.f, 0.f, 0.f, 0.f, 0.f, 0.f, 0.f, 0.f};

  // ---- phase 1: S tile rows (16 x 1024), each wave does a 16-col slice ----
  const unsigned short* qrow = qkv + (size_t)(b * NN + i0 + lr) * DQKV + h * DHEAD;
  for (int jt = 0; jt < NN; jt += 64) {
    const int j = jt + wave * 16;
    v8f acc = z;
#pragma unroll
    for (int kk = 0; kk < DHEAD; kk += 32) {
      Frag a;
#pragma unroll
      for (int v = 0; v < 8; ++v)
        a.u[v] = *(const unsigned*)(qrow + kk + koB(v, hl));
      Frag bfr;
      const unsigned short* kp =
          qkv + (size_t)(b * NN + j + lr) * DQKV + DQ + h * DHEAD + kk;
#pragma unroll
      for (int v = 0; v < 8; ++v)
        bfr.u[v] = *(const unsigned*)(kp + koB(v, hl));
      acc = __builtin_amdgcn_wmma_f32_16x16x32_bf16(
          false, a.v, false, bfr.v, (short)0, acc, false, false);
    }
    const float k2n = k2[(size_t)bh * NN + j + lr];
#pragma unroll
    for (int r = 0; r < 8; ++r) {
      const int m = r + 8 * hl;
      float d2 = q2[(size_t)bh * NN + i0 + m] + k2n - 2.0f * acc[r];
      d2 = fmaxf(d2, 0.0f);
      S[m][j + lr] = f2bf(__expf(-d2 * inv_s2));
    }
  }
  __syncthreads();

  // ---- phase 2: out(16x512) = S(16x1024) @ V(1024x512); wave -> 128 cols ---
  v8f acc[8] = {z, z, z, z, z, z, z, z};
  const int c0 = wave * 128;
  const unsigned short* vbase = qkv + (size_t)(b * NN) * DQKV + 2 * DQ + h * DIN;
  for (int kk = 0; kk < NN; kk += 32) {
    Frag a;
#pragma unroll
    for (int v = 0; v < 8; ++v)
      a.u[v] = *(const unsigned*)&S[lr][kk + koB(v, hl)];
#pragma unroll
    for (int t = 0; t < 8; ++t) {
      Frag bfr;
      const int c = c0 + t * 16 + lr;
      const unsigned short* vp = vbase + (size_t)(kk + 16 * hl) * DQKV + c;
#pragma unroll
      for (int v = 0; v < 8; ++v) {
        unsigned lo = vp[(size_t)(2 * v) * DQKV];
        unsigned hi = vp[(size_t)(2 * v + 1) * DQKV];
        bfr.u[v] = lo | (hi << 16);
      }
      acc[t] = __builtin_amdgcn_wmma_f32_16x16x32_bf16(
          false, a.v, false, bfr.v, (short)0, acc[t], false, false);
    }
  }
#pragma unroll
  for (int t = 0; t < 8; ++t) {
#pragma unroll
    for (int r = 0; r < 8; ++r) {
      const int row = i0 + r + 8 * hl;
      outm[(size_t)(b * NN + row) * DV + h * DIN + c0 + t * 16 + lr] =
          f2bf(acc[t][r]);
    }
  }
}

// ---------------------------------------------------------------------------
// Host-side orchestration
// ---------------------------------------------------------------------------
extern "C" void kernel_launch(void* const* d_in, const int* in_sizes, int n_in,
                              void* d_out, int out_size, void* d_ws, size_t ws_size,
                              hipStream_t stream) {
  (void)in_sizes; (void)n_in; (void)out_size; (void)ws_size;
  const float* x     = (const float*)d_in[0];
  const float* pos   = (const float*)d_in[1];
  const float* g1    = (const float*)d_in[2];
  const float* be1   = (const float*)d_in[3];
  const float* scale = (const float*)d_in[4];
  const float* Wqkv  = (const float*)d_in[5];
  const float* bqkv  = (const float*)d_in[6];
  const float* Wm    = (const float*)d_in[7];
  const float* bm    = (const float*)d_in[8];
  const float* g2    = (const float*)d_in[9];
  const float* be2   = (const float*)d_in[10];
  const float* W1    = (const float*)d_in[11];
  const float* b1    = (const float*)d_in[12];
  const float* g3    = (const float*)d_in[13];
  const float* be3   = (const float*)d_in[14];
  const float* W2    = (const float*)d_in[15];
  const float* b2    = (const float*)d_in[16];
  float* out = (float*)d_out;

  char* w = (char*)d_ws;
  auto alloc = [&](size_t bytes) {
    char* p = w;
    w += (bytes + 255) & ~(size_t)255;
    return p;
  };
  unsigned short* act1    = (unsigned short*)alloc((size_t)MROWS * DIN * 2);
  unsigned short* Wqkv_t  = (unsigned short*)alloc((size_t)DQKV * DIN * 2);
  unsigned short* Wm_t    = (unsigned short*)alloc((size_t)DIN * DV * 2);
  unsigned short* W1_t    = (unsigned short*)alloc((size_t)DEXP * DIN * 2);
  unsigned short* W2_t    = (unsigned short*)alloc((size_t)DIN * DEXP * 2);
  unsigned short* qkvb    = (unsigned short*)alloc((size_t)MROWS * DQKV * 2);
  float*          q2      = (float*)alloc((size_t)BB * HH * NN * 4);
  float*          k2      = (float*)alloc((size_t)BB * HH * NN * 4);
  unsigned short* attn_o  = (unsigned short*)alloc((size_t)MROWS * DV * 2);
  float*          x2      = (float*)alloc((size_t)MROWS * DIN * 4);
  unsigned short* n2      = (unsigned short*)alloc((size_t)MROWS * DIN * 2);
  float*          ffh     = (float*)alloc((size_t)MROWS * DEXP * 4);
  unsigned short* act2    = (unsigned short*)alloc((size_t)MROWS * DEXP * 2);

  // weights -> bf16, transposed (NxK)
  wtrans_kernel<<<(DIN * DQKV + 255) / 256, 256, 0, stream>>>(Wqkv, Wqkv_t, DIN, DQKV);
  wtrans_kernel<<<(DV * DIN + 255) / 256, 256, 0, stream>>>(Wm, Wm_t, DV, DIN);
  wtrans_kernel<<<(DIN * DEXP + 255) / 256, 256, 0, stream>>>(W1, W1_t, DIN, DEXP);
  wtrans_kernel<<<(DEXP * DIN + 255) / 256, 256, 0, stream>>>(W2, W2_t, DEXP, DIN);

  // stage 1: act1 = swish(featnorm(x) + pos)
  featnorm_kernel<<<BB * DIN, 256, 0, stream>>>(x, g1, be1, pos, act1, DIN, 1);

  // stage 2: qkv = act1 @ Wqkv + b_qkv  (bf16 out)
  gemm_bf16_kernel<<<dim3(DQKV / 128, MROWS / 128), 256, 0, stream>>>(
      act1, Wqkv_t, bqkv, nullptr, nullptr, qkvb, MROWS, DQKV, DIN);

  // stage 3: attention (RBF kernel scores, no softmax)
  qk2_kernel<<<(BB * HH * NN) / 256, 256, 0, stream>>>(qkvb, q2, k2);
  attn_kernel<<<BB * HH * (NN / 16), 128, 0, stream>>>(qkvb, q2, k2, attn_o, scale);

  // stage 4: x2 = x + attn_o @ Wm + bm
  gemm_bf16_kernel<<<dim3(DIN / 128, MROWS / 128), 256, 0, stream>>>(
      attn_o, Wm_t, bm, x, x2, nullptr, MROWS, DIN, DV);

  // stage 5: ffh = featnorm(x2) @ W1 + b1
  featnorm_kernel<<<BB * DIN, 256, 0, stream>>>(x2, g2, be2, nullptr, n2, DIN, 0);
  gemm_bf16_kernel<<<dim3(DEXP / 128, MROWS / 128), 256, 0, stream>>>(
      n2, W1_t, b1, nullptr, ffh, nullptr, MROWS, DEXP, DIN);

  // stage 6: act2 = swish(featnorm(ffh));  out = act2 @ W2 + b2 + x2
  featnorm_kernel<<<BB * DEXP, 256, 0, stream>>>(ffh, g3, be3, nullptr, act2, DEXP, 1);
  gemm_bf16_kernel<<<dim3(DIN / 128, MROWS / 128), 256, 0, stream>>>(
      act2, W2_t, b2, x2, out, nullptr, MROWS, DIN, DEXP);
}
